// NSDE_43628277792779
// MI455X (gfx1250) — matrix operand order repaired and checked by reading, not compiled, not run on hardware
//
#include <hip/hip_runtime.h>
#include <hip/hip_bf16.h>

typedef _Float16 v16h __attribute__((ext_vector_type(16)));
typedef _Float16 v8h  __attribute__((ext_vector_type(8)));
typedef _Float16 v2h  __attribute__((ext_vector_type(2)));
typedef float    v8f  __attribute__((ext_vector_type(8)));

#define N_PATHS 512
#define N_STEPS 32
#define B_OPTS  64
#define HID     64
#define V0_INIT 0.2f
#define NWG     (N_PATHS / 2)   // 256 workgroups, 2 paths each

// Hardware transcendental tanh on gfx1250 (single v_tanh_f32, no libm branch
// cascade); TRANS32 ops co-execute with the WMMA/XDL pipe.
__device__ __forceinline__ float fast_tanh(float x) {
#if __has_builtin(__builtin_amdgcn_tanhf)
    return __builtin_amdgcn_tanhf(x);
#else
    return tanhf(x);
#endif
}

__device__ __forceinline__ float dot2_acc(v2h a, v2h b, float acc) {
#if __has_builtin(__builtin_amdgcn_fdot2)
    return __builtin_amdgcn_fdot2(a, b, acc, false);
#else
    return acc + (float)a[0] * (float)b[0] + (float)a[1] * (float)b[1];
#endif
}

// 256 threads = 8 wave32; wave w owns rows 16w..16w+15 of the 128-row
// (2 paths x 64 options) tile. All weights resident in LDS as f16.
__global__ __launch_bounds__(256)
void nsde_mc_kernel(const float* __restrict__ S0,
                    const float* __restrict__ Kstrike,
                    const float* __restrict__ Tmat,
                    const float* __restrict__ rf_ptr,
                    const float* __restrict__ Z1,
                    const float* __restrict__ Z2,
                    const float* __restrict__ W0,    // (4,4,64)
                    const float* __restrict__ b0,    // (4,64)
                    const float* __restrict__ W1,    // (4,64,64)
                    const float* __restrict__ b1,    // (4,64)
                    const float* __restrict__ W2,    // (4,64,1)
                    const float* __restrict__ b2,    // (4,1)
                    float* __restrict__ partials)    // (NWG, 64)
{
    __shared__ __align__(16) _Float16 sW0T[4][64][4];   // [net][n][k]  (B-frag friendly)
    __shared__ __align__(32) _Float16 sW1T[4][64][64];  // [net][n][k]  (transposed)
    __shared__ float    sB0[4][64];
    __shared__ float    sB1[4][64];
    __shared__ __align__(16) _Float16 sW2h[4][64];      // f16 for packed fdot2
    __shared__ float    sB2[4];
    __shared__ __align__(16) _Float16 sH[8][16 * 64];   // per-wave activation tile
    __shared__ float    sS[128], sV[128], sDt[128];
    __shared__ float    sPart[2][64];

    const int tid = threadIdx.x;
    const int wg  = blockIdx.x;

    // ---- stage weights into LDS (f16, transposed for B-fragment reads) ----
    for (int i = tid; i < 4 * 64 * 64; i += 256) {
        int g = i >> 12, rem = i & 4095, n = rem >> 6, k = rem & 63;
        sW1T[g][n][k] = (_Float16)W1[g * 4096 + k * 64 + n];
    }
    for (int i = tid; i < 4 * 64 * 4; i += 256) {
        int g = i >> 8, rem = i & 255, n = rem >> 2, k = rem & 3;
        sW0T[g][n][k] = (_Float16)W0[g * 256 + k * 64 + n];
    }
    {
        int g = tid >> 6, n = tid & 63;
        sB0[g][n]  = b0[tid];
        sB1[g][n]  = b1[tid];
        sW2h[g][n] = (_Float16)W2[tid];
    }
    if (tid < 4) sB2[tid] = b2[tid];

    const float rf = rf_ptr[0];

    if (tid < 128) {
        int o = tid & 63;
        sS[tid]  = S0[o];
        sV[tid]  = V0_INIT;
        sDt[tid] = Tmat[o] / (float)N_STEPS;
    }
    __syncthreads();

    const int wave   = tid >> 5;
    const int lane   = tid & 31;
    const int M      = lane & 15;        // A-row index == B/C column index
    const int h      = lane >> 4;        // half-wave selector in fragment layouts
    const int row    = wave * 16 + M;    // 0..127
    const int opt    = row & 63;
    const int plocal = row >> 6;
    const int path   = wg * 2 + plocal;
    _Float16* hbuf   = &sH[wave][0];

    const float dt   = sDt[row];
    const float sqdt = sqrtf(dt);

    __builtin_prefetch(&Z1[path * N_STEPS], 0, 0);
    __builtin_prefetch(&Z2[path * N_STEPS], 0, 0);

    // Evaluate MLP net g on the layer-0 A fragment a0; returns scalar output
    // for this lane's row M (valid in both half-waves after the xor-reduce).
    auto mlp = [&](int g, v16h a0) -> float {
        // ---- layer 0: 16x64 = 4 WMMA tiles, K padded 4 -> 32 ----
        for (int nt = 0; nt < 4; ++nt) {
            const int N = nt * 16 + M;
            v16h b = {};
            _Float16 w0 = sW0T[g][N][0], w1 = sW0T[g][N][1];
            _Float16 w2 = sW0T[g][N][2], w3 = sW0T[g][N][3];
            b[0] = (h == 0) ? w0 : (_Float16)0.0f;   // K=0..3 live only in h==0
            b[1] = (h == 0) ? w1 : (_Float16)0.0f;
            b[2] = (h == 0) ? w2 : (_Float16)0.0f;
            b[3] = (h == 0) ? w3 : (_Float16)0.0f;
            const float bias = sB0[g][N];
            v8f c;
            #pragma unroll
            for (int r = 0; r < 8; ++r) c[r] = bias;
            v8f d = __builtin_amdgcn_wmma_f32_16x16x32_f16(
                        false, a0, false, b, (short)0, c, false, false);
            #pragma unroll
            for (int r = 0; r < 8; ++r)
                hbuf[(r + 8 * h) * 64 + N] = (_Float16)fast_tanh(d[r]);
        }
        // ---- reload h0 as layer-1 A fragments (K=64 -> two K32 chunks) ----
        v16h a1[2];
        #pragma unroll
        for (int kc = 0; kc < 2; ++kc) {
            const v8h lo = *(const v8h*)&hbuf[M * 64 + 32 * kc + 8 * h];
            const v8h hi = *(const v8h*)&hbuf[M * 64 + 32 * kc + 16 + 8 * h];
            #pragma unroll
            for (int i = 0; i < 8; ++i) { a1[kc][i] = lo[i]; a1[kc][8 + i] = hi[i]; }
        }
        // ---- layer 1: 2 K-chunks x 4 N-tiles = 8 WMMA ----
        for (int nt = 0; nt < 4; ++nt) {
            const int N = nt * 16 + M;
            const float bias = sB1[g][N];
            v8f c;
            #pragma unroll
            for (int r = 0; r < 8; ++r) c[r] = bias;
            #pragma unroll
            for (int kc = 0; kc < 2; ++kc) {
                const v8h blo = *(const v8h*)&sW1T[g][N][32 * kc + 8 * h];
                const v8h bhi = *(const v8h*)&sW1T[g][N][32 * kc + 16 + 8 * h];
                v16h bw;
                #pragma unroll
                for (int i = 0; i < 8; ++i) { bw[i] = blo[i]; bw[8 + i] = bhi[i]; }
                c = __builtin_amdgcn_wmma_f32_16x16x32_f16(
                        false, a1[kc], false, bw, (short)0, c, false, false);
            }
            #pragma unroll
            for (int r = 0; r < 8; ++r)
                hbuf[(r + 8 * h) * 64 + N] = (_Float16)fast_tanh(c[r]);
        }
        // ---- layer 2: per-row dot(h1, W2) via packed f16 dot2, f32 acc;
        //      split across the two half-waves, combined with one xor-shuffle.
        float acc = 0.0f;
        const v2h* hp = (const v2h*)&hbuf[M * 64 + 32 * h];
        const v2h* wp = (const v2h*)&sW2h[g][32 * h];
        #pragma unroll
        for (int n = 0; n < 16; ++n)
            acc = dot2_acc(hp[n], wp[n], acc);
        acc += __shfl_xor(acc, 16, 32);
        return acc + sB2[g];
    };

    for (int j = 0; j < N_STEPS; ++j) {
        const float t  = dt * (float)j;
        const float z1 = Z1[path * N_STEPS + j];
        const float z2 = Z2[path * N_STEPS + j];

        const float Sv = sS[row];
        const float Vv = sV[row];

        v16h a0 = {};
        a0[0] = (h == 0) ? (_Float16)Sv : (_Float16)0.0f;  // K=0..3 in h==0 half
        a0[1] = (h == 0) ? (_Float16)Vv : (_Float16)0.0f;
        a0[2] = (h == 0) ? (_Float16)rf : (_Float16)0.0f;
        a0[3] = (h == 0) ? (_Float16)t  : (_Float16)0.0f;

        const float f0 = mlp(0, a0);
        const float f1 = mlp(1, a0);
        const float Snew = Sv * (1.0f + f0 * dt + f1 * z1);
        if (h == 0) sS[row] = Snew;

        v16h a2 = {};
        a2[0] = (h == 0) ? (_Float16)Snew : (_Float16)0.0f; // uses UPDATED S
        a2[1] = (h == 0) ? (_Float16)Vv   : (_Float16)0.0f;
        a2[2] = (h == 0) ? (_Float16)rf   : (_Float16)0.0f;
        a2[3] = (h == 0) ? (_Float16)t    : (_Float16)0.0f;

        const float f2 = mlp(2, a2);
        const float f3 = mlp(3, a2);
        if (h == 0) sV[row] = Vv * (1.0f + f2 * dt + f3 * sqdt * z2);
    }

    // ---- payoff and deterministic per-block partial sums ----
    if (h == 0) {
        const float Sq  = sS[row];
        const float pay = (Sq - Kstrike[opt] < 0.0f) ? 0.0f : Sq;
        sPart[plocal][opt] = expf(-rf * Tmat[opt]) * pay * (1.0f / (float)N_PATHS);
    }
    __syncthreads();
    if (tid < 64) partials[wg * 64 + tid] = sPart[0][tid] + sPart[1][tid];
}

// Fixed-order reduction over workgroup partials -> bit-deterministic output.
__global__ __launch_bounds__(64)
void nsde_reduce_kernel(const float* __restrict__ partials,
                        float* __restrict__ out)
{
    const int t = threadIdx.x;
    float s = 0.0f;
    for (int w = 0; w < NWG; ++w) s += partials[w * 64 + t];
    out[t] = s;
}

extern "C" void kernel_launch(void* const* d_in, const int* in_sizes, int n_in,
                              void* d_out, int out_size, void* d_ws, size_t ws_size,
                              hipStream_t stream) {
    const float* S0 = (const float*)d_in[0];
    const float* K  = (const float*)d_in[1];
    const float* T  = (const float*)d_in[2];
    const float* rf = (const float*)d_in[3];
    const float* Z1 = (const float*)d_in[4];
    const float* Z2 = (const float*)d_in[5];
    const float* W0 = (const float*)d_in[6];
    const float* b0 = (const float*)d_in[7];
    const float* W1 = (const float*)d_in[8];
    const float* b1 = (const float*)d_in[9];
    const float* W2 = (const float*)d_in[10];
    const float* b2 = (const float*)d_in[11];

    float* partials = (float*)d_ws;            // NWG*64 floats = 64 KB
    float* out      = (float*)d_out;           // 64 floats

    nsde_mc_kernel<<<NWG, 256, 0, stream>>>(S0, K, T, rf, Z1, Z2,
                                            W0, b0, W1, b1, W2, b2, partials);
    nsde_reduce_kernel<<<1, 64, 0, stream>>>(partials, out);
}
